// GATs_76476187672732
// MI455X (gfx1250) — compile-verified, matrix operand order checked by reading, not run on hardware
//
#include <hip/hip_runtime.h>
#include <hip/hip_bf16.h>
#include <stdint.h>

// ---------------------------------------------------------------------------
// GAT x4 (GATConv + GraphNorm + PReLU) for gfx1250 (MI455X).
// GEMMs: v_wmma_f32_16x16x32_bf16, double-buffered LDS staging via
// global_load_async_to_lds_b128 (ASYNCcnt pipeline). Softmax/scatter via
// global atomics (edge working sets stay resident in 192MB L2).
// ---------------------------------------------------------------------------

typedef __attribute__((ext_vector_type(16))) __bf16 v16bf;
typedef __attribute__((ext_vector_type(8)))  __bf16 v8bf;
typedef __attribute__((ext_vector_type(8)))  float  v8f;

#define NODES   16384
#define HC      256
#define HEADS   4
#define CHN     64
#define GN_EPS  1e-5f
#define NEG_SLOPE 0.2f

__device__ __forceinline__ __bf16 f2bf(float f) {
  union { float f; unsigned u; } in; in.f = f;
  unsigned u = in.u;
  u += 0x7FFFu + ((u >> 16) & 1u);            // round-to-nearest-even
  union { unsigned short s; __bf16 b; } out;
  out.s = (unsigned short)(u >> 16);
  return out.b;
}

// monotone float <-> uint encoding for atomic max over floats
__device__ __forceinline__ unsigned fenc(float f) {
  unsigned u = __float_as_uint(f);
  return (u >> 31) ? ~u : (u | 0x80000000u);
}
__device__ __forceinline__ float fdec(unsigned u) {
  return (u >> 31) ? __uint_as_float(u ^ 0x80000000u) : __uint_as_float(~u);
}

// ------------------------- conversion kernels ------------------------------

__global__ void k_cvt_x(const float* __restrict__ x, __bf16* __restrict__ xb, int n) {
  int i = blockIdx.x * blockDim.x + threadIdx.x;
  if (i < n) xb[i] = f2bf(x[i]);
}

// W [K, 256] fp32 -> Wt [256, K] bf16 (transposed so B fragments load contiguously)
__global__ void k_cvt_wt(const float* __restrict__ W, __bf16* __restrict__ Wt, int K) {
  int i = blockIdx.x * blockDim.x + threadIdx.x;
  if (i >= 256 * K) return;
  int n = i / K, k = i - n * K;
  Wt[i] = f2bf(W[k * 256 + n]);
}

// ------------------------------ WMMA GEMM ----------------------------------
// C[M,256] = A[M,K] (bf16) x Wt^T (Wt stored [256,K] bf16), fp32 accumulate.
// Block (8 waves) computes 128 rows x 64 cols. Double-buffered pipeline:
//   stage tile i+1 (async DMA to LDS) || compute tile i (ds_load + 4 WMMA),
//   then s_wait_asynccnt 0 + barrier.
#define A_ELEMS (128 * 32)
#define B_ELEMS (64 * 32)

__global__ void __launch_bounds__(256) k_gemm_wmma(const __bf16* __restrict__ A,
                                                   const __bf16* __restrict__ Bt,
                                                   float* __restrict__ Cmat, int K) {
  __shared__ __bf16 smA[2 * A_ELEMS];
  __shared__ __bf16 smB[2 * B_ELEMS];
  const int t    = threadIdx.x;
  const int lane = t & 31;
  const int wid  = t >> 5;
  const int row0 = blockIdx.y * 128;
  const int col0 = blockIdx.x * 64;
  const int r    = lane & 15;
  const int kb   = (lane < 16) ? 0 : 8;        // K sub-block per ISA A/B layout

  const unsigned aBase = (unsigned)(uintptr_t)(&smA[0]);  // LDS byte offset
  const unsigned bBase = (unsigned)(uintptr_t)(&smB[0]);

  // per-thread staging sources (16B chunks); advance 64B per K-step
  const int ca0 = t, ca1 = t + 256;            // A: 512 chunks of 16B
  const char* ga0 = (const char*)(A + (size_t)(row0 + (ca0 >> 2)) * K) + (ca0 & 3) * 16;
  const char* ga1 = (const char*)(A + (size_t)(row0 + (ca1 >> 2)) * K) + (ca1 & 3) * 16;
  const char* gb  = (const char*)(Bt + (size_t)(col0 + (t >> 2)) * K) + (t & 3) * 16;
  const unsigned la0 = aBase + (unsigned)ca0 * 16u;
  const unsigned la1 = aBase + (unsigned)ca1 * 16u;
  const unsigned lb  = bBase + (unsigned)t * 16u;

  auto stage = [&](int k0, int buf) {
    const unsigned ao = (unsigned)buf * (A_ELEMS * 2u);   // bytes
    const unsigned bo = (unsigned)buf * (B_ELEMS * 2u);
    asm volatile("global_load_async_to_lds_b128 %0, %1, off"
                 :: "v"(la0 + ao), "v"(ga0 + (size_t)k0 * 2) : "memory");
    asm volatile("global_load_async_to_lds_b128 %0, %1, off"
                 :: "v"(la1 + ao), "v"(ga1 + (size_t)k0 * 2) : "memory");
    asm volatile("global_load_async_to_lds_b128 %0, %1, off"
                 :: "v"(lb + bo),  "v"(gb  + (size_t)k0 * 2) : "memory");
  };

  v8f acc[4] = {{}, {}, {}, {}};
  const int nk = K >> 5;

  stage(0, 0);
  asm volatile("s_wait_asynccnt 0x0" ::: "memory");
  __syncthreads();

  for (int i = 0; i < nk; ++i) {
    const int cur = i & 1;
    if (i + 1 < nk) stage((i + 1) << 5, cur ^ 1);   // prefetch next tile

    const __bf16* sa = &smA[cur * A_ELEMS];
    const __bf16* sb = &smB[cur * B_ELEMS];
    union BV { v16bf v; v8bf h[2]; } ua, ub[4];
    const __bf16* ap = sa + (wid * 16 + r) * 32 + kb;
    ua.h[0] = *(const v8bf*)(ap);
    ua.h[1] = *(const v8bf*)(ap + 16);
#pragma unroll
    for (int nt = 0; nt < 4; ++nt) {
      const __bf16* bp = sb + (nt * 16 + r) * 32 + kb;
      ub[nt].h[0] = *(const v8bf*)(bp);
      ub[nt].h[1] = *(const v8bf*)(bp + 16);
    }
#pragma unroll
    for (int nt = 0; nt < 4; ++nt) {
      acc[nt] = __builtin_amdgcn_wmma_f32_16x16x32_bf16(
          false, ua.v, false, ub[nt].v, (short)0, acc[nt], false, false);
    }

    asm volatile("s_wait_asynccnt 0x0" ::: "memory");  // next-tile DMA landed
    __syncthreads();                                   // all reads of cur done
  }

  // D layout: VGPR v -> M = v (lanes 0-15) or v+8 (lanes 16-31); N = lane&15
  const int mb = (lane < 16) ? 0 : 8;
#pragma unroll
  for (int nt = 0; nt < 4; ++nt) {
    float* cp = Cmat + (size_t)(row0 + wid * 16 + mb) * HC + col0 + nt * 16 + r;
#pragma unroll
    for (int v = 0; v < 8; ++v) cp[(size_t)v * HC] = acc[nt][v];
  }
}

// --------------------- attention coefficients per node ---------------------
// a_src[n,h] = dot(xl[n,h,:], att_src[h,:]); one wave per node, 8 ch per lane.
__global__ void k_att(const float* __restrict__ xl, const float* __restrict__ att_s,
                      const float* __restrict__ att_d, float* __restrict__ asrc,
                      float* __restrict__ adst) {
  int lane = threadIdx.x & 31;
  int node = blockIdx.x * 8 + (threadIdx.x >> 5);
  const float* xr = xl + (size_t)node * HC + lane * 8;
  const float* as = att_s + lane * 8;
  const float* ad = att_d + lane * 8;
  float ps = 0.f, pd = 0.f;
#pragma unroll
  for (int j = 0; j < 8; ++j) { float v = xr[j]; ps += v * as[j]; pd += v * ad[j]; }
#pragma unroll
  for (int off = 1; off < 8; off <<= 1) {
    ps += __shfl_xor(ps, off, 32);
    pd += __shfl_xor(pd, off, 32);
  }
  if ((lane & 7) == 0) {
    int h = lane >> 3;
    asrc[node * HEADS + h] = ps;
    adst[node * HEADS + h] = pd;
  }
}

// --------------------------- edge construction -----------------------------
__global__ void k_edges(const int* __restrict__ ei, int* __restrict__ src2,
                        int* __restrict__ dst2, int E, int E2) {
  int i = blockIdx.x * blockDim.x + threadIdx.x;
  if (i >= E2) return;
  if (i < E) { src2[i] = ei[i]; dst2[i] = ei[E + i]; }
  else       { src2[i] = i - E; dst2[i] = i - E; }
}

// pass 1: e = leakyrelu(a_src[src]+a_dst[dst]); segment max via atomic
__global__ void k_edge_max(const int* __restrict__ src2, const int* __restrict__ dst2,
                           const float* __restrict__ asrc, const float* __restrict__ adst,
                           float* __restrict__ ebuf, unsigned* __restrict__ menc, int EH) {
  int i = blockIdx.x * blockDim.x + threadIdx.x;
  if (i >= EH) return;
  int e = i >> 2, h = i & 3;
  float v = asrc[src2[e] * 4 + h] + adst[dst2[e] * 4 + h];
  v = v > 0.f ? v : NEG_SLOPE * v;
  ebuf[i] = v;
  atomicMax(&menc[dst2[e] * 4 + h], fenc(v));
}

// pass 2: e = exp(e - m[dst]); segment sum via atomic
__global__ void k_edge_exp(const int* __restrict__ dst2, float* __restrict__ ebuf,
                           const unsigned* __restrict__ menc, float* __restrict__ sbuf, int EH) {
  int i = blockIdx.x * blockDim.x + threadIdx.x;
  if (i >= EH) return;
  int e = i >> 2, h = i & 3;
  int d = dst2[e];
  float ex = __expf(ebuf[i] - fdec(menc[d * 4 + h]));
  ebuf[i] = ex;
  atomicAdd(&sbuf[d * 4 + h], ex);
}

__global__ void k_bias_init(float* __restrict__ outp, const float* __restrict__ bias, int n) {
  int i = blockIdx.x * blockDim.x + threadIdx.x;
  if (i < n) outp[i] = bias[i & (HC - 1)];
}

// pass 3: alpha = e/(s+eps); out[dst] += alpha * xl[src]. 64 threads per edge.
__global__ void k_agg(const int* __restrict__ src2, const int* __restrict__ dst2,
                      const float* __restrict__ ebuf, const float* __restrict__ sbuf,
                      const float* __restrict__ xl, float* __restrict__ outp,
                      float* __restrict__ alphab, int E2) {
  int t = blockIdx.x * blockDim.x + threadIdx.x;
  int e = t >> 6;
  int c = t & 63;
  if (e >= E2) return;
  int s = src2[e], d = dst2[e];
#pragma unroll
  for (int h = 0; h < HEADS; ++h) {
    float al = ebuf[e * 4 + h] / (sbuf[d * 4 + h] + 1e-16f);
    if (c == h) alphab[e * 4 + h] = al;
    int ch = h * CHN + c;
    atomicAdd(&outp[(size_t)d * HC + ch], al * xl[(size_t)s * HC + ch]);
  }
}

// ------------------------------ GraphNorm ----------------------------------
__global__ void k_colsum(const float* __restrict__ x, float* __restrict__ acc, int npb) {
  int ch = threadIdx.x;                        // 256
  int n0 = blockIdx.x * npb;
  float s = 0.f;
  for (int n = n0; n < n0 + npb; ++n) s += x[(size_t)n * HC + ch];
  atomicAdd(&acc[ch], s);
}

__global__ void k_varsum(const float* __restrict__ x, const float* __restrict__ suma,
                         const float* __restrict__ gms, float* __restrict__ vacc, int npb) {
  int ch = threadIdx.x;
  int n0 = blockIdx.x * npb;
  float msc = suma[ch] * (1.f / NODES) * gms[ch];
  float s = 0.f;
  for (int n = n0; n < n0 + npb; ++n) {
    float c = x[(size_t)n * HC + ch] - msc;
    s += c * c;
  }
  atomicAdd(&vacc[ch], s);
}

__global__ void k_norm_prelu(const float* __restrict__ x, const float* __restrict__ suma,
                             const float* __restrict__ vacc, const float* __restrict__ gms,
                             const float* __restrict__ gw, const float* __restrict__ gb,
                             const float* __restrict__ prelu, float* __restrict__ xn, int n) {
  int i = blockIdx.x * blockDim.x + threadIdx.x;
  if (i >= n) return;
  int ch = i & (HC - 1);
  float msc = suma[ch] * (1.f / NODES) * gms[ch];
  float c = x[i] - msc;
  float var = vacc[ch] * (1.f / NODES);
  float y = gw[ch] * c * __frsqrt_rn(var + GN_EPS) + gb[ch];
  float p = prelu[0];
  xn[i] = (y >= 0.f) ? y : p * y;
}

// ---------------------------------------------------------------------------

extern "C" void kernel_launch(void* const* d_in, const int* in_sizes, int n_in,
                              void* d_out, int out_size, void* d_ws, size_t ws_size,
                              hipStream_t stream) {
  const int N  = NODES;
  const int E  = in_sizes[1] / 2;
  const int E2 = E + N;
  const int EH = E2 * HEADS;

  const float* x_in = (const float*)d_in[0];
  const int*   ei   = (const int*)d_in[1];

  // bump allocator over d_ws (256B aligned)
  char* base = (char*)d_ws;
  size_t off = 0;
  auto alloc = [&](size_t bytes) -> void* {
    void* p = base + off;
    off += (bytes + 255) & ~(size_t)255;
    return p;
  };
  float*    xA    = (float*)alloc((size_t)N * HC * 4);
  float*    xB    = (float*)alloc((size_t)N * HC * 4);
  float*    xl    = (float*)alloc((size_t)N * HC * 4);
  float*    agg   = (float*)alloc((size_t)N * HC * 4);
  __bf16*   xbf   = (__bf16*)alloc((size_t)N * 256 * 2);
  __bf16*   wtbf  = (__bf16*)alloc((size_t)256 * 256 * 2);
  float*    asrc  = (float*)alloc((size_t)N * HEADS * 4);
  float*    adst  = (float*)alloc((size_t)N * HEADS * 4);
  int*      src2  = (int*)alloc((size_t)E2 * 4);
  int*      dst2  = (int*)alloc((size_t)E2 * 4);
  float*    ebuf  = (float*)alloc((size_t)EH * 4);
  float*    alphab= (float*)alloc((size_t)EH * 4);
  unsigned* menc  = (unsigned*)alloc((size_t)N * HEADS * 4);
  float*    sbuf  = (float*)alloc((size_t)N * HEADS * 4);
  float*    suma  = (float*)alloc(256 * 4);
  float*    vacc  = (float*)alloc(256 * 4);

  // edge list with self loops (deterministic, rebuilt every call)
  k_edges<<<(E2 + 255) / 256, 256, 0, stream>>>(ei, src2, dst2, E, E2);

  const float* xcur = x_in;
  for (int layer = 0; layer < 4; ++layer) {
    const int pb = 2 + 8 * layer;
    const float* W     = (const float*)d_in[pb + 0];
    const float* att_s = (const float*)d_in[pb + 1];
    const float* att_d = (const float*)d_in[pb + 2];
    const float* bias  = (const float*)d_in[pb + 3];
    const float* gw    = (const float*)d_in[pb + 4];
    const float* gb    = (const float*)d_in[pb + 5];
    const float* gms   = (const float*)d_in[pb + 6];
    const float* prelu = (const float*)d_in[pb + 7];
    const int K = (layer == 0) ? 128 : 256;
    float* xnext = (layer % 2 == 0) ? xA : xB;

    // convert inputs to bf16
    k_cvt_x<<<((size_t)N * K + 255) / 256, 256, 0, stream>>>(xcur, xbf, N * K);
    k_cvt_wt<<<((size_t)256 * K + 255) / 256, 256, 0, stream>>>(W, wtbf, K);

    // xl = x @ W  (WMMA bf16, fp32 accumulate) : 128x64 macro-tiles
    dim3 ggrid(256 / 64, N / 128);
    k_gemm_wmma<<<ggrid, 256, 0, stream>>>(xbf, wtbf, xl, K);

    // attention coefficients
    k_att<<<N / 8, 256, 0, stream>>>(xl, att_s, att_d, asrc, adst);

    // segment softmax over edges
    hipMemsetAsync(menc, 0, (size_t)N * HEADS * 4, stream);
    hipMemsetAsync(sbuf, 0, (size_t)N * HEADS * 4, stream);
    k_edge_max<<<(EH + 255) / 256, 256, 0, stream>>>(src2, dst2, asrc, adst, ebuf, menc, EH);
    k_edge_exp<<<(EH + 255) / 256, 256, 0, stream>>>(dst2, ebuf, menc, sbuf, EH);

    // aggregate: out = bias + sum alpha * xl[src]
    k_bias_init<<<((size_t)N * HC + 255) / 256, 256, 0, stream>>>(agg, bias, N * HC);
    k_agg<<<((size_t)E2 * 64 + 255) / 256, 256, 0, stream>>>(src2, dst2, ebuf, sbuf, xl, agg,
                                                             alphab, E2);

    // GraphNorm + PReLU
    hipMemsetAsync(suma, 0, 256 * 4, stream);
    hipMemsetAsync(vacc, 0, 256 * 4, stream);
    k_colsum<<<64, 256, 0, stream>>>(agg, suma, N / 64);
    k_varsum<<<64, 256, 0, stream>>>(agg, suma, gms, vacc, N / 64);
    k_norm_prelu<<<((size_t)N * HC + 255) / 256, 256, 0, stream>>>(
        agg, suma, vacc, gms, gw, gb, prelu, xnext, N * HC);

    xcur = xnext;
  }

  // Outputs: x [N*256] f32, stack([src,dst]) [2*E2] i32, alpha [E2*4] f32
  hipMemcpyAsync(d_out, xcur, (size_t)N * HC * 4, hipMemcpyDeviceToDevice, stream);
  int* oi = (int*)d_out + (size_t)N * HC;
  hipMemcpyAsync(oi, src2, (size_t)E2 * 4, hipMemcpyDeviceToDevice, stream);
  hipMemcpyAsync(oi + E2, dst2, (size_t)E2 * 4, hipMemcpyDeviceToDevice, stream);
  float* oa = (float*)d_out + (size_t)N * HC + 2 * (size_t)E2;
  hipMemcpyAsync(oa, alphab, (size_t)EH * 4, hipMemcpyDeviceToDevice, stream);
}
